// RNNDecoder_63032940036358
// MI455X (gfx1250) — matrix operand order repaired and checked by reading, not compiled, not run on hardware
//
#include <hip/hip_runtime.h>
#include <hip/hip_bf16.h>
#include <math.h>

typedef __attribute__((ext_vector_type(16))) __bf16 v16bf;
typedef __attribute__((ext_vector_type(8)))  __bf16 v8bf;
typedef __attribute__((ext_vector_type(8)))  float  v8f;

#define B_   256
#define L_   400
#define V_   50000
#define E_   512
#define H_   1024
#define MEM_ 1024
#define OOV_ 10
#define VO_  (V_ + OOV_)   // 50010

// ---------------------------------------------------------------------------
// bf16-WMMA GEMM: C[M=256, N] = A[256, K] * W[N, K]^T + bias[N]
// Block = 256 threads = 8 waves. Each wave: one 16-col N strip x all 256 rows.
// Double-buffered LDS A slab (2 x 256x32 bf16); next-slab global loads are
// issued before the WMMA phase and converted/stored after it. A-fragment LDS
// loads are software-pipelined one M-tile ahead of the WMMA that consumes them.
// Requires: K % 32 == 0, N % 16 == 0.
// ---------------------------------------------------------------------------
__global__ __launch_bounds__(256) void k_gemm_bf16(
    const float* __restrict__ A, const float* __restrict__ W,
    const float* __restrict__ bias, float* __restrict__ C, int N, int K)
{
    __shared__ __bf16 As[2][256 * 32];   // 2 x 16 KB
    const int tid  = threadIdx.x;
    const int wave = tid >> 5;
    const int lane = tid & 31;
    const int g    = lane >> 4;          // half-wave select
    const int lr   = lane & 15;
    const int ntile = blockIdx.x * 8 + wave;
    const int n     = ntile * 16 + lr;            // this lane's output column
    const bool active = (ntile * 16) < N;         // wave-uniform -> EXEC all-1 at WMMA

    v8f acc[16] = {};

    // ---- prologue: stage slab 0
    {
        const float* ar = A + (size_t)tid * K;
        __bf16* dst = &As[0][tid * 32];
        #pragma unroll
        for (int j = 0; j < 32; j += 4) {
            float4 v = *(const float4*)(ar + j);
            dst[j + 0] = (__bf16)v.x; dst[j + 1] = (__bf16)v.y;
            dst[j + 2] = (__bf16)v.z; dst[j + 3] = (__bf16)v.w;
        }
    }

    int p = 0;
    for (int k0 = 0; k0 < K; k0 += 32, p ^= 1) {
        __syncthreads();   // slab p staged and visible; prev reads of p^1 done

        // issue W loads for this k-step (waited on just before use)
        float4 w0, w1, w2, w3;
        if (active) {
            const float* wr = W + (size_t)n * K + k0 + 16 * g;
            w0 = *(const float4*)(wr + 0);
            w1 = *(const float4*)(wr + 4);
            w2 = *(const float4*)(wr + 8);
            w3 = *(const float4*)(wr + 12);
        }

        // issue next-slab global loads now; convert/store after the WMMA phase
        const bool more = (k0 + 32) < K;
        float4 st[8];
        if (more) {
            const float* ar = A + (size_t)tid * K + k0 + 32;
            #pragma unroll
            for (int j = 0; j < 8; ++j) st[j] = *(const float4*)(ar + 4 * j);
        }

        if (active) {
            v16bf bf;
            bf[0]  = (__bf16)w0.x; bf[1]  = (__bf16)w0.y; bf[2]  = (__bf16)w0.z; bf[3]  = (__bf16)w0.w;
            bf[4]  = (__bf16)w1.x; bf[5]  = (__bf16)w1.y; bf[6]  = (__bf16)w1.z; bf[7]  = (__bf16)w1.w;
            bf[8]  = (__bf16)w2.x; bf[9]  = (__bf16)w2.y; bf[10] = (__bf16)w2.z; bf[11] = (__bf16)w2.w;
            bf[12] = (__bf16)w3.x; bf[13] = (__bf16)w3.y; bf[14] = (__bf16)w3.z; bf[15] = (__bf16)w3.w;

            const __bf16* base = As[p];
            // pipeline: fragment for tile mt+1 loads while tile mt does WMMA
            v8bf c0 = *(const v8bf*)(base + lr * 32 + 16 * g);
            v8bf c1 = *(const v8bf*)(base + lr * 32 + 8 + 16 * g);
            #pragma unroll
            for (int mt = 0; mt < 16; ++mt) {
                v16bf af;
                #pragma unroll
                for (int j = 0; j < 8; ++j) { af[j] = c0[j]; af[j + 8] = c1[j]; }
                if (mt < 15) {
                    const __bf16* arow = base + ((mt + 1) * 16 + lr) * 32;
                    c0 = *(const v8bf*)(arow + 16 * g);
                    c1 = *(const v8bf*)(arow + 8 + 16 * g);
                }
                acc[mt] = __builtin_amdgcn_wmma_f32_16x16x32_bf16(
                    false, af, false, bf, (short)0, acc[mt], false, false);
            }
        }

        if (more) {
            __bf16* dst = &As[p ^ 1][tid * 32];
            #pragma unroll
            for (int j = 0; j < 8; ++j) {
                dst[4 * j + 0] = (__bf16)st[j].x; dst[4 * j + 1] = (__bf16)st[j].y;
                dst[4 * j + 2] = (__bf16)st[j].z; dst[4 * j + 3] = (__bf16)st[j].w;
            }
        }
    }

    if (active) {
        const float bv = bias ? bias[n] : 0.0f;
        #pragma unroll
        for (int mt = 0; mt < 16; ++mt) {
            #pragma unroll
            for (int r = 0; r < 8; ++r) {
                const int m = mt * 16 + g * 8 + r;   // C/D f32 layout: M = r + 8g (+16*mt)
                C[(size_t)m * N + n] = acc[mt][r] + bv;
            }
        }
    }
}

// ---------------------------------------------------------------------------
// y_emb gather + x = concat(y_emb, context)
// ---------------------------------------------------------------------------
__global__ void k_embed_concat(const int* __restrict__ y,
                               const float* __restrict__ emb,
                               const float* __restrict__ context,
                               float* __restrict__ yemb, float* __restrict__ x)
{
    int idx = blockIdx.x * blockDim.x + threadIdx.x;       // B_*(E_+MEM_)
    if (idx >= B_ * (E_ + MEM_)) return;
    int b = idx / (E_ + MEM_), j = idx % (E_ + MEM_);
    float v;
    if (j < E_) { v = emb[(size_t)y[b] * E_ + j]; yemb[b * E_ + j] = v; }
    else        { v = context[b * MEM_ + (j - E_)]; }
    x[idx] = v;
}

// ---------------------------------------------------------------------------
// GRU gates (PyTorch order r,z,n): h1 = (1-z)*n + z*h0
// ---------------------------------------------------------------------------
__global__ void k_gru_gates(const float* __restrict__ gi, const float* __restrict__ gh,
                            const float* __restrict__ h0,
                            float* __restrict__ h1, float* __restrict__ hnext)
{
    int idx = blockIdx.x * blockDim.x + threadIdx.x;       // B_*H_
    if (idx >= B_ * H_) return;
    int b = idx / H_, j = idx % H_;
    const float* gib = gi + (size_t)b * 3 * H_;
    const float* ghb = gh + (size_t)b * 3 * H_;
    float r = 1.f / (1.f + expf(-(gib[j] + ghb[j])));
    float z = 1.f / (1.f + expf(-(gib[H_ + j] + ghb[H_ + j])));
    float n = tanhf(gib[2 * H_ + j] + r * ghb[2 * H_ + j]);
    float h = (1.f - z) * n + z * h0[idx];
    h1[idx] = h; hnext[idx] = h;
}

// ---------------------------------------------------------------------------
// Fused attention (flash-style, single HBM pass over memory_bank):
//   scores -> online softmax -> ctx, plus attn written at the end.
// Block per batch row: 8 waves; wave w scores row l0+w (wave-reduced dot),
// then all 256 threads update their 4-wide ctx slice for those 8 rows
// (rows are hot in WGP$/L2 from the score pass). Raw masked scores are kept
// in LDS so attn can be emitted with the final (m, s).
// ---------------------------------------------------------------------------
__global__ __launch_bounds__(256) void k_attn_fused(
    const float* __restrict__ bank, const float* __restrict__ dec,
    const unsigned char* __restrict__ mask,
    float* __restrict__ attn, float* __restrict__ ctx)
{
    __shared__ float fs[MEM_];   // dec_feat row
    __shared__ float sc[L_];     // raw masked scores
    __shared__ float scw[8];     // current chunk scores
    const int b = blockIdx.x, tid = threadIdx.x;
    const int wave = tid >> 5, lane = tid & 31;

    for (int k = tid; k < MEM_; k += 256) fs[k] = dec[(size_t)b * MEM_ + k];
    __syncthreads();

    float m_run = -INFINITY, s_run = 0.f;
    float4 cacc = make_float4(0.f, 0.f, 0.f, 0.f);
    const int m0 = tid * 4;

    for (int l0 = 0; l0 < L_; l0 += 8) {          // L_ % 8 == 0
        const int l = l0 + wave;
        {   // wave-parallel score for row l
            const float* r = bank + ((size_t)b * L_ + l) * MEM_;
            float s = 0.f;
            for (int k = lane; k < MEM_; k += 32) s += r[k] * fs[k];
            for (int off = 16; off > 0; off >>= 1) s += __shfl_xor(s, off, 32);
            if (lane == 0) {
                float sv = mask[b * L_ + l] ? s : -1e9f;
                scw[wave] = sv; sc[l] = sv;
            }
        }
        __syncthreads();

        // redundant identical update of running stats on every thread
        float mchunk = scw[0];
        #pragma unroll
        for (int w = 1; w < 8; ++w) mchunk = fmaxf(mchunk, scw[w]);
        float m_new = fmaxf(m_run, mchunk);
        float alpha = expf(m_run - m_new);
        float p[8], psum = 0.f;
        #pragma unroll
        for (int w = 0; w < 8; ++w) { p[w] = expf(scw[w] - m_new); psum += p[w]; }
        s_run = s_run * alpha + psum;
        m_run = m_new;
        cacc.x *= alpha; cacc.y *= alpha; cacc.z *= alpha; cacc.w *= alpha;

        #pragma unroll
        for (int w = 0; w < 8; ++w) {
            float4 v = *(const float4*)(bank + ((size_t)b * L_ + l0 + w) * MEM_ + m0);
            cacc.x += p[w] * v.x; cacc.y += p[w] * v.y;
            cacc.z += p[w] * v.z; cacc.w += p[w] * v.w;
        }
        __syncthreads();   // protect scw before next chunk
    }

    const float inv = 1.f / s_run;
    float4 o = make_float4(cacc.x * inv, cacc.y * inv, cacc.z * inv, cacc.w * inv);
    *(float4*)(ctx + (size_t)b * MEM_ + m0) = o;
    for (int l = tid; l < L_; l += 256)
        attn[(size_t)b * L_ + l] = expf(sc[l] - m_run) * inv;
}

// vd_in = concat(ctx, h1)
__global__ void k_concat_vd(const float* __restrict__ ctx, const float* __restrict__ h1,
                            float* __restrict__ vdin)
{
    int idx = blockIdx.x * blockDim.x + threadIdx.x;       // B_*2048
    if (idx >= B_ * (MEM_ + H_)) return;
    int b = idx / (MEM_ + H_), j = idx % (MEM_ + H_);
    vdin[idx] = (j < MEM_) ? ctx[b * MEM_ + j] : h1[b * H_ + (j - MEM_)];
}

// p_gen[b] = sigmoid(W_pgen . [ctx, h1, y_emb] + b_pgen); one wave per batch row
__global__ __launch_bounds__(256) void k_pgen(const float* __restrict__ ctx,
                                              const float* __restrict__ h1,
                                              const float* __restrict__ yemb,
                                              const float* __restrict__ Wp,
                                              const float* __restrict__ bp,
                                              float* __restrict__ out)
{
    int b    = (blockIdx.x * 256 + threadIdx.x) >> 5;
    int lane = threadIdx.x & 31;
    if (b >= B_) return;
    float s = 0.f;
    for (int k = lane; k < MEM_; k += 32) s += Wp[k] * ctx[(size_t)b * MEM_ + k];
    for (int k = lane; k < H_;   k += 32) s += Wp[MEM_ + k] * h1[(size_t)b * H_ + k];
    for (int k = lane; k < E_;   k += 32) s += Wp[MEM_ + H_ + k] * yemb[(size_t)b * E_ + k];
    for (int off = 16; off > 0; off >>= 1) s += __shfl_xor(s, off, 32);
    if (lane == 0) out[b] = 1.f / (1.f + expf(-(s + bp[0])));
}

// ---------------------------------------------------------------------------
// softmax over V=50000, scaled by p_gen, written into final_dist; zero OOV tail
// ---------------------------------------------------------------------------
__global__ __launch_bounds__(256) void k_vocab_final(const float* __restrict__ logits,
                                                     const float* __restrict__ pgen,
                                                     float* __restrict__ out)
{
    __shared__ float sm_m[256], sm_s[256];
    int b = blockIdx.x, tid = threadIdx.x;
    const float* row = logits + (size_t)b * V_;
    float mx = -INFINITY, sm = 0.f;
    for (int v = tid; v < V_; v += 256) {         // online max/sum
        float x = row[v];
        float m2 = fmaxf(mx, x);
        sm = sm * expf(mx - m2) + expf(x - m2);
        mx = m2;
    }
    sm_m[tid] = mx; sm_s[tid] = sm; __syncthreads();
    for (int off = 128; off > 0; off >>= 1) {
        if (tid < off) {
            float m1 = sm_m[tid], s1 = sm_s[tid];
            float m2 = sm_m[tid + off], s2 = sm_s[tid + off];
            float m = fmaxf(m1, m2);
            sm_m[tid] = m;
            sm_s[tid] = s1 * expf(m1 - m) + s2 * expf(m2 - m);
        }
        __syncthreads();
    }
    float M = sm_m[0];
    float scale = pgen[b] / sm_s[0];
    float* orow = out + (size_t)b * VO_;
    for (int v = tid; v < V_; v += 256) orow[v] = expf(row[v] - M) * scale;
    for (int v = V_ + tid; v < VO_; v += 256) orow[v] = 0.f;
}

// scatter-add the copy distribution
__global__ void k_scatter(const int* __restrict__ src_oov,
                          const float* __restrict__ attn,
                          const float* __restrict__ pgen,
                          float* __restrict__ final_dist)
{
    int idx = blockIdx.x * blockDim.x + threadIdx.x;       // B_*L_
    if (idx >= B_ * L_) return;
    int b = idx / L_;
    float w = (1.f - pgen[b]) * attn[idx];
    atomicAdd(final_dist + (size_t)b * VO_ + src_oov[idx], w);
}

__global__ void k_copy(const float* __restrict__ src, float* __restrict__ dst, int n)
{
    int idx = blockIdx.x * blockDim.x + threadIdx.x;
    if (idx < n) dst[idx] = src[idx];
}

// ---------------------------------------------------------------------------
extern "C" void kernel_launch(void* const* d_in, const int* in_sizes, int n_in,
                              void* d_out, int out_size, void* d_ws, size_t ws_size,
                              hipStream_t stream)
{
    const int*   y       = (const int*)  d_in[0];
    const float* h0      = (const float*)d_in[1];   // [1,B,H] == [B,H]
    const float* bank    = (const float*)d_in[2];   // [B,L,MEM]
    const unsigned char* mask = (const unsigned char*)d_in[3]; // bool [B,L]
    const float* context = (const float*)d_in[4];
    const int*   src_oov = (const int*)  d_in[6];
    const float* coverage= (const float*)d_in[7];
    const float* emb     = (const float*)d_in[8];
    const float* W_ih    = (const float*)d_in[9];
    const float* W_hh    = (const float*)d_in[10];
    const float* b_ih    = (const float*)d_in[11];
    const float* b_hh    = (const float*)d_in[12];
    const float* W_attn  = (const float*)d_in[13];
    const float* b_attn  = (const float*)d_in[14];
    const float* W_pgen  = (const float*)d_in[15];
    const float* b_pgen  = (const float*)d_in[16];
    const float* W_v1    = (const float*)d_in[17];
    const float* b_v1    = (const float*)d_in[18];
    const float* W_v2    = (const float*)d_in[19];
    const float* b_v2    = (const float*)d_in[20];

    // ---- workspace layout (floats)
    float* ws = (float*)d_ws;
    float* yemb   = ws;                               // B*E
    float* x      = yemb   + (size_t)B_ * E_;         // B*1536
    float* gi     = x      + (size_t)B_ * (E_ + MEM_);// B*3072
    float* gh     = gi     + (size_t)B_ * 3 * H_;     // B*3072
    float* h1     = gh     + (size_t)B_ * 3 * H_;     // B*H
    float* dec    = h1     + (size_t)B_ * H_;         // B*MEM
    float* vdin   = dec    + (size_t)B_ * MEM_;       // B*2048
    float* v1o    = vdin   + (size_t)B_ * (MEM_ + H_);// B*H
    float* logits = v1o    + (size_t)B_ * H_;         // B*V

    // ---- output layout (floats, concatenated in return order)
    float* out        = (float*)d_out;
    float* o_final    = out;
    float* o_hnext    = o_final + (size_t)B_ * VO_;
    float* o_ctx      = o_hnext + (size_t)B_ * H_;
    float* o_attn     = o_ctx   + (size_t)B_ * MEM_;
    float* o_pgen     = o_attn  + (size_t)B_ * L_;
    float* o_cov      = o_pgen  + (size_t)B_;

    // 1) embedding gather + input concat
    k_embed_concat<<<(B_ * (E_ + MEM_) + 255) / 256, 256, 0, stream>>>(y, emb, context, yemb, x);

    // 2) GRU input/hidden GEMMs (bf16 WMMA)
    k_gemm_bf16<<<(3 * H_ / 16 + 7) / 8, 256, 0, stream>>>(x,  W_ih, b_ih, gi, 3 * H_, E_ + MEM_);
    k_gemm_bf16<<<(3 * H_ / 16 + 7) / 8, 256, 0, stream>>>(h0, W_hh, b_hh, gh, 3 * H_, H_);

    // 3) gates -> h1 (ws) and h_next (out)
    k_gru_gates<<<(B_ * H_ + 255) / 256, 256, 0, stream>>>(gi, gh, h0, h1, o_hnext);

    // 4) attention features + fused scores/softmax/ctx (single bank pass)
    k_gemm_bf16<<<(MEM_ / 16 + 7) / 8, 256, 0, stream>>>(h1, W_attn, b_attn, dec, MEM_, H_);
    k_attn_fused<<<B_, 256, 0, stream>>>(bank, dec, mask, o_attn, o_ctx);

    // 5) vocab path: v1 then v2 (big WMMA GEMM), p_gen
    k_concat_vd<<<(B_ * (MEM_ + H_) + 255) / 256, 256, 0, stream>>>(o_ctx, h1, vdin);
    k_gemm_bf16<<<(H_ / 16 + 7) / 8, 256, 0, stream>>>(vdin, W_v1, b_v1, v1o, H_, MEM_ + H_);
    k_gemm_bf16<<<(V_ / 16 + 7) / 8, 256, 0, stream>>>(v1o, W_v2, b_v2, logits, V_, H_);
    k_pgen<<<B_ / 8, 256, 0, stream>>>(o_ctx, h1, yemb, W_pgen, b_pgen, o_pgen);

    // 6) final distribution: scaled softmax + zero OOV tail, then scatter-add
    k_vocab_final<<<B_, 256, 0, stream>>>(logits, o_pgen, o_final);
    k_scatter<<<(B_ * L_ + 255) / 256, 256, 0, stream>>>(src_oov, o_attn, o_pgen, o_final);

    // 7) coverage passthrough
    k_copy<<<(B_ * L_ + 255) / 256, 256, 0, stream>>>(coverage, o_cov, B_ * L_);
}
